// GCN2_12412455486113
// MI455X (gfx1250) — matrix operand order repaired and checked by reading, not compiled, not run on hardware
//
#include <hip/hip_runtime.h>
#include <cmath>

typedef __attribute__((ext_vector_type(16))) _Float16 v16h;
typedef __attribute__((ext_vector_type(8)))  _Float16 v8h;
typedef __attribute__((ext_vector_type(8)))  float    v8f;

#define HIDC 96
#define ALPHA_C 0.1f
#define BN_EPS_C 1e-5f

// ---------------- elementwise / utility kernels ----------------

__global__ void k_fill_f32(float* __restrict__ p, float v, int n) {
  int i = blockIdx.x * blockDim.x + threadIdx.x;
  if (i < n) p[i] = v;
}

__global__ void k_zero_f32(float* __restrict__ p, int n) {
  int i = blockIdx.x * blockDim.x + threadIdx.x;
  int stride = gridDim.x * blockDim.x;
  for (; i < n; i += stride) p[i] = 0.f;
}

__global__ void k_deg_accum(const int* __restrict__ col, float* __restrict__ deg, int E) {
  int i = blockIdx.x * blockDim.x + threadIdx.x;
  if (i < E) atomicAdd(&deg[col[i]], 1.f);
}

__global__ void k_dinv(float* __restrict__ deg, int n) {
  int i = blockIdx.x * blockDim.x + threadIdx.x;
  if (i < n) {
    float d = deg[i];
    deg[i] = (d > 0.f) ? rsqrtf(d) : 0.f;
  }
}

__global__ void k_edge_w(const int* __restrict__ row, const int* __restrict__ col,
                         const float* __restrict__ dinv, float* __restrict__ ew, int E) {
  int i = blockIdx.x * blockDim.x + threadIdx.x;
  if (i < E) ew[i] = dinv[row[i]] * dinv[col[i]];
}

// ---------------- f16 packing (pad + transpose) ----------------

// x [M x Kin] f32 -> d [M x Kp] f16 zero-padded (Kp power of two)
__global__ void k_pack_x(const float* __restrict__ s, _Float16* __restrict__ d,
                         int total, int Kin, int Kp) {
  int i = blockIdx.x * blockDim.x + threadIdx.x;
  if (i >= total) return;
  int m = i / Kp, k = i - m * Kp;
  d[i] = (k < Kin) ? (_Float16)s[(size_t)m * Kin + k] : (_Float16)0.f;
}

// W0 [Kin x 96] -> BT [96 x Kp] f16 (transposed, zero-padded K)
__global__ void k_pack_w0t(const float* __restrict__ s, _Float16* __restrict__ d,
                           int Kin, int Kp) {
  int i = blockIdx.x * blockDim.x + threadIdx.x;
  if (i >= HIDC * Kp) return;
  int n = i / Kp, k = i - n * Kp;
  d[i] = (k < Kin) ? (_Float16)s[(size_t)k * HIDC + n] : (_Float16)0.f;
}

// Ws [8 x 96 x 96] -> BT per layer [8 x 96 x 96] f16 (each layer transposed)
__global__ void k_pack_wst(const float* __restrict__ s, _Float16* __restrict__ d, int total) {
  int i = blockIdx.x * blockDim.x + threadIdx.x;
  if (i >= total) return;
  int l = i / (HIDC * HIDC);
  int r = i - l * (HIDC * HIDC);
  int n = r / HIDC, k = r - n * HIDC;
  d[i] = (_Float16)s[(size_t)l * HIDC * HIDC + (size_t)k * HIDC + n];
}

// W_out [96 x 40] -> BT [48 x 96] f16 (transposed, zero-padded N)
__global__ void k_pack_woutt(const float* __restrict__ s, _Float16* __restrict__ d) {
  int i = blockIdx.x * blockDim.x + threadIdx.x;
  if (i >= 48 * HIDC) return;
  int n = i / HIDC, k = i - n * HIDC;
  d[i] = (n < 40) ? (_Float16)s[(size_t)k * 40 + n] : (_Float16)0.f;
}

// ---------------- sparse aggregation ----------------
// block = (96, 4): threadIdx.x = channel, threadIdx.y = edge-in-block
__global__ void k_scatter(const int* __restrict__ row, const int* __restrict__ col,
                          const float* __restrict__ ew, const _Float16* __restrict__ h16,
                          float* __restrict__ agg, int E) {
  int e = blockIdx.x * blockDim.y + threadIdx.y;
  if (e >= E) return;
  int c = threadIdx.x;
  int r = row[e], t = col[e];
  float w = ew[e];
  float v = w * (float)h16[(size_t)r * HIDC + c];
  atomicAdd(&agg[(size_t)t * HIDC + c], v);
}

// z = (1-alpha)*(agg + dinv^2*h) + alpha*x0 ; zf written in place over agg; f16 over h16
__global__ void k_zcomb(float* __restrict__ agg, const float* __restrict__ dinv,
                        const float* __restrict__ x0, _Float16* __restrict__ h16, int NH) {
  int i = blockIdx.x * blockDim.x + threadIdx.x;
  if (i >= NH) return;
  int n = i / HIDC;
  float di = dinv[n];
  float hv = (float)h16[i];
  float z = (1.f - ALPHA_C) * (agg[i] + di * di * hv) + ALPHA_C * x0[i];
  agg[i] = z;
  h16[i] = (_Float16)z;
}

// ---------------- BatchNorm ----------------

__global__ void k_bn_stats(const float* __restrict__ d, float* __restrict__ gsum,
                           float* __restrict__ gsq, int total) {
  __shared__ float s_sum[HIDC];
  __shared__ float s_sq[HIDC];
  for (int c = threadIdx.x; c < HIDC; c += blockDim.x) { s_sum[c] = 0.f; s_sq[c] = 0.f; }
  __syncthreads();
  for (int i = blockIdx.x * blockDim.x + threadIdx.x; i < total; i += gridDim.x * blockDim.x) {
    float v = d[i];
    int c = i % HIDC;
    atomicAdd(&s_sum[c], v);
    atomicAdd(&s_sq[c], v * v);
  }
  __syncthreads();
  for (int c = threadIdx.x; c < HIDC; c += blockDim.x) {
    atomicAdd(&gsum[c], s_sum[c]);
    atomicAdd(&gsq[c], s_sq[c]);
  }
}

__global__ void k_bn_finalize(const float* __restrict__ gsum, const float* __restrict__ gsq,
                              const float* __restrict__ gamma, const float* __restrict__ beta,
                              float* __restrict__ scale, float* __restrict__ shift, float invN) {
  int c = blockIdx.x * blockDim.x + threadIdx.x;
  if (c >= HIDC) return;
  float mean = gsum[c] * invN;
  float var  = gsq[c] * invN - mean * mean;
  float inv  = rsqrtf(var + BN_EPS_C);
  float sc = gamma[c] * inv;
  scale[c] = sc;
  shift[c] = beta[c] - mean * sc;
}

// h = relu(optional BN(d)); writes f16 h, optionally f32 x0 copy
__global__ void k_apply_act(const float* __restrict__ d, const float* __restrict__ scale,
                            const float* __restrict__ shift, _Float16* __restrict__ h16,
                            float* __restrict__ x0, int NH, int use_bn) {
  int i = blockIdx.x * blockDim.x + threadIdx.x;
  if (i >= NH) return;
  float v = d[i];
  if (use_bn) {
    int c = i % HIDC;
    v = v * scale[c] + shift[c];
  }
  v = fmaxf(v, 0.f);
  h16[i] = (_Float16)v;
  if (x0) x0[i] = v;
}

// ---------------- WMMA GEMM ----------------
// One wave = one 16x16 C tile; blockDim = 256 (8 waves). K is compile-time,
// multiple of 32, so the chunk loop fully unrolls into b128 loads + v_wmma.
// A: [M x K] f16 row-major. BT: [N16 x K] f16 (B transposed).
// mode 0: D = A@B + bias[n]     mode 1: D = c1*Z[m*N16+n] + c2*(A@B)
template <int K>
__global__ void k_wmma_gemm(const _Float16* __restrict__ A, const _Float16* __restrict__ BT,
                            const float* __restrict__ Z, const float* __restrict__ bias,
                            float* __restrict__ D,
                            int M, int N16, int Nstore, int ldd,
                            float c1, float c2, int mode) {
  const int wid = blockIdx.x * (blockDim.x >> 5) + (threadIdx.x >> 5);
  const int ntiles = N16 >> 4;
  const int mt = wid / ntiles;
  const int nt = wid - mt * ntiles;
  if (mt * 16 >= M) return;
  const int lane = threadIdx.x & 31;
  const int lr = lane & 15;
  const int hf = lane >> 4;
  int m = mt * 16 + lr;
  if (m >= M) m = M - 1;              // clamp load row (stores guarded below)
  const int n = nt * 16 + lr;
  const _Float16* arow = A + (size_t)m * K;   // lane's A row
  const _Float16* brow = BT + (size_t)n * K;  // lane's B column (contiguous in BT)

  v8f acc = {};
#pragma unroll
  for (int k0 = 0; k0 < K; k0 += 32) {
    // A frag (ISA 7.12.2): a[t]   = A[m][k0 + hf*8 + t],      t = 0..7
    //                      a[8+t] = A[m][k0 + 16 + hf*8 + t]
    v8h alo = *(const v8h*)(arow + k0 + hf * 8);
    v8h ahi = *(const v8h*)(arow + k0 + 16 + hf * 8);
    // B frag: b[t] = BT[n][k0 + hf*8 + t], b[8+t] = BT[n][k0 + 16 + hf*8 + t], t = 0..7
    v8h blo = *(const v8h*)(brow + k0 + hf * 8);
    v8h bhi = *(const v8h*)(brow + k0 + 16 + hf * 8);
    v16h a = __builtin_shufflevector(alo, ahi, 0, 1, 2, 3, 4, 5, 6, 7,
                                     8, 9, 10, 11, 12, 13, 14, 15);
    v16h b = __builtin_shufflevector(blo, bhi, 0, 1, 2, 3, 4, 5, 6, 7,
                                     8, 9, 10, 11, 12, 13, 14, 15);
    acc = __builtin_amdgcn_wmma_f32_16x16x32_f16(false, a, false, b, (short)0, acc,
                                                 false, false);
  }

#pragma unroll
  for (int r = 0; r < 8; ++r) {
    // C layout: VGPR r -> M = r + 8*hf ; N = lane%16
    const int mr = mt * 16 + hf * 8 + r;
    const int nc = nt * 16 + lr;
    if (mr < M && nc < Nstore) {
      float v = acc[r];
      if (mode == 0) v += bias[nc];
      else v = c1 * Z[(size_t)mr * N16 + nc] + c2 * v;
      D[(size_t)mr * ldd + nc] = v;
    }
  }
}

// ---------------- host orchestration ----------------

extern "C" void kernel_launch(void* const* d_in, const int* in_sizes, int n_in,
                              void* d_out, int out_size, void* d_ws, size_t ws_size,
                              hipStream_t stream) {
  const float* x        = (const float*)d_in[0];
  const int*   ei       = (const int*)d_in[1];
  const float* W0       = (const float*)d_in[2];
  const float* b0       = (const float*)d_in[3];
  const float* Ws       = (const float*)d_in[4];
  const float* bn_gamma = (const float*)d_in[5];
  const float* bn_beta  = (const float*)d_in[6];
  const float* W_out    = (const float*)d_in[7];
  const float* b_out    = (const float*)d_in[8];
  float* out = (float*)d_out;

  const int IN_DIM = 500, KP = 512, NL = 8, OUTD = 40, OUTP = 48;
  const int N  = in_sizes[0] / IN_DIM;
  const int E  = in_sizes[1] / 2;
  const int NH = N * HIDC;
  const int* row = ei;
  const int* col = ei + E;

  // carve workspace (256B-aligned slices)
  char* p = (char*)d_ws;
  auto carve = [&](size_t bytes) -> void* {
    void* r = (void*)p;
    p += (bytes + 255) & ~(size_t)255;
    return r;
  };
  float*    dinv   = (float*)carve((size_t)N * 4);
  float*    ew     = (float*)carve((size_t)E * 4);
  _Float16* x16    = (_Float16*)carve((size_t)N * KP * 2);   // padded f16 x
  _Float16* h16    = (_Float16*)carve((size_t)NH * 2);
  float*    x0     = (float*)carve((size_t)NH * 4);
  float*    agg    = (float*)carve((size_t)NH * 4);          // also holds z (f32)
  float*    df     = (float*)carve((size_t)NH * 4);
  _Float16* w0t    = (_Float16*)carve((size_t)HIDC * KP * 2);
  _Float16* wst    = (_Float16*)carve((size_t)NL * HIDC * HIDC * 2);
  _Float16* woutt  = (_Float16*)carve((size_t)OUTP * HIDC * 2);
  float*    bnsum  = (float*)carve(2 * HIDC * 4);  // sum | sumsq
  float*    bnsq   = bnsum + HIDC;
  float*    scale  = (float*)carve(2 * HIDC * 4);  // scale | shift
  float*    shift  = scale + HIDC;

  const int T = 256;
  auto cdiv = [](int a, int b) { return (a + b - 1) / b; };

  // --- gcn_norm: deg (self-loop = 1) -> dinv -> edge weights ---
  k_fill_f32<<<cdiv(N, T), T, 0, stream>>>(dinv, 1.f, N);
  k_deg_accum<<<cdiv(E, T), T, 0, stream>>>(col, dinv, E);
  k_dinv<<<cdiv(N, T), T, 0, stream>>>(dinv, N);
  k_edge_w<<<cdiv(E, T), T, 0, stream>>>(row, col, dinv, ew, E);

  // --- pack inputs/weights into f16 (transposed B layouts, padded K/N) ---
  k_pack_x<<<cdiv(N * KP, T), T, 0, stream>>>(x, x16, N * KP, IN_DIM, KP);
  k_pack_w0t<<<cdiv(HIDC * KP, T), T, 0, stream>>>(W0, w0t, IN_DIM, KP);
  k_pack_wst<<<cdiv(NL * HIDC * HIDC, T), T, 0, stream>>>(Ws, wst, NL * HIDC * HIDC);
  k_pack_woutt<<<cdiv(OUTP * HIDC, T), T, 0, stream>>>(W_out, woutt);

  const int mtiles = cdiv(N, 16);
  auto gemm_blocks = [&](int n16) { return cdiv(mtiles * (n16 >> 4), 8); };

  // --- input layer: df = x @ W0 + b0 ; BN ; ReLU -> h16, x0 ---
  k_wmma_gemm<KP><<<gemm_blocks(HIDC), T, 0, stream>>>(
      x16, w0t, nullptr, b0, df, N, HIDC, HIDC, HIDC, 0.f, 0.f, 0);
  k_zero_f32<<<1, T, 0, stream>>>(bnsum, 2 * HIDC);
  k_bn_stats<<<512, T, 0, stream>>>(df, bnsum, bnsq, NH);
  k_bn_finalize<<<1, 128, 0, stream>>>(bnsum, bnsq, bn_gamma, bn_beta, scale, shift,
                                       1.f / (float)N);
  k_apply_act<<<cdiv(NH, T), T, 0, stream>>>(df, scale, shift, h16, x0, NH, 1);

  // --- GCN2 layers ---
  for (int i = 1; i <= NL; ++i) {
    const float bet = logf(0.5f / (float)i + 1.f);

    k_zero_f32<<<cdiv(NH, T * 4), T, 0, stream>>>(agg, NH);
    {
      dim3 blk(HIDC, 4);
      k_scatter<<<cdiv(E, 4), blk, 0, stream>>>(row, col, ew, h16, agg, E);
    }
    k_zcomb<<<cdiv(NH, T), T, 0, stream>>>(agg, dinv, x0, h16, NH);

    // df = (1-bet)*z + bet*(z @ Ws[i-1]) ; z f32 in agg, z f16 in h16
    k_wmma_gemm<HIDC><<<gemm_blocks(HIDC), T, 0, stream>>>(
        h16, wst + (size_t)(i - 1) * HIDC * HIDC, agg, nullptr, df,
        N, HIDC, HIDC, HIDC, 1.f - bet, bet, 1);

    const int use_bn = (i != NL - 1) ? 1 : 0;
    if (use_bn) {
      k_zero_f32<<<1, T, 0, stream>>>(bnsum, 2 * HIDC);
      k_bn_stats<<<512, T, 0, stream>>>(df, bnsum, bnsq, NH);
      k_bn_finalize<<<1, 128, 0, stream>>>(bnsum, bnsq, bn_gamma + (size_t)i * HIDC,
                                           bn_beta + (size_t)i * HIDC, scale, shift,
                                           1.f / (float)N);
    }
    k_apply_act<<<cdiv(NH, T), T, 0, stream>>>(df, scale, shift, h16, nullptr, NH, use_bn);
  }

  // --- output: out = h @ W_out + b_out (N padded to 48, store 40) ---
  k_wmma_gemm<HIDC><<<gemm_blocks(OUTP), T, 0, stream>>>(
      h16, woutt, nullptr, b_out, out, N, OUTP, OUTD, OUTD, 0.f, 0.f, 0);
}
